// TinyEmbedFFNN_2327872274769
// MI455X (gfx1250) — matrix operand order repaired and checked by reading, not compiled, not run on hardware
//
#include <hip/hip_runtime.h>
#include <math.h>

typedef __attribute__((ext_vector_type(16))) _Float16 v16h;
typedef __attribute__((ext_vector_type(8)))  _Float16 v8h;
typedef __attribute__((ext_vector_type(8)))  float    v8f;

#define BDIM   256        // batch
#define PDIM   100        // positions
#define EDIM   32         // embedding width
#define VDIM   1000       // vocab
#define HDIM   256        // hidden
#define NFEAT  33         // embedded features
#define KDIM   1120       // 35 * 32
#define KSLICE 35         // K / 32
#define NTILES 16         // H / 16
#define TOKTILE 16        // tokens per block (WMMA M)
#define ROWSTR 1128       // padded LDS row stride in halves (16B-aligned, conflict-spread)

// ---------------------------------------------------------------------------
// Kernel 1: pack W1 (H x K, f32, row-major) into f16 WMMA B-fragments.
// Fragment layout this code uses with V_WMMA_F32_16X16X32_F16 (wave32; contiguous K, same as A):
//   lane l -> column n = l & 15 ; lanes 0-15 hold K=kb..kb+15, lanes 16-31
//   hold K=kb+16..kb+31 ; 16 halves per lane packed 2-per-dword (even K low).
// Buffer index: [(s*NTILES + t)*32 + l] * 16 halves.
// ---------------------------------------------------------------------------
__global__ __launch_bounds__(256)
void pack_w1_kernel(const float* __restrict__ W1, _Float16* __restrict__ bpack)
{
    int tid = blockIdx.x * blockDim.x + threadIdx.x;   // exactly KSLICE*NTILES*32 threads
    int s   = tid / (NTILES * 32);
    int rem = tid % (NTILES * 32);
    int t   = rem >> 5;
    int l   = rem & 31;
    int n   = t * 16 + (l & 15);
    int kb  = s * 32 + ((l < 16) ? 0 : 16);
    const float* src = W1 + (size_t)n * KDIM + kb;
    v16h frag;
#pragma unroll
    for (int j = 0; j < 16; ++j) frag[j] = (_Float16)src[j];
    ((v16h*)bpack)[(size_t)(s * NTILES + t) * 32 + l] = frag;
}

// combine two 8-half LDS loads into one 16-half A fragment
__device__ __forceinline__ v16h comb(v8h lo, v8h hi)
{
    v16h a;
#pragma unroll
    for (int i = 0; i < 8; ++i) { a[i] = lo[i]; a[8 + i] = hi[i]; }
    return a;
}

// ---------------------------------------------------------------------------
// Kernel 2: fused feature-gather + WMMA GEMM + bias/ReLU + W2 dot -> logit.
// Block = 256 threads = 8 wave32s; wave w owns N-tiles {2w, 2w+1}; block owns
// 16 tokens. K loop is double-buffered with two explicit live buffer sets
// (cannot be coalesced away), so LDS/global loads overlap the WMMA chain.
// ---------------------------------------------------------------------------
__global__ __launch_bounds__(256)
void embed_ffnn_kernel(const float* __restrict__ x,       // (B*P, 35)
                       const float* __restrict__ tables,  // (NF, V, E)
                       const _Float16* __restrict__ bpack,
                       const float* __restrict__ b1,      // (H)
                       const float* __restrict__ W2,      // (H)
                       const float* __restrict__ b2,      // (1)
                       float* __restrict__ logits)        // (B*P)
{
    __shared__ _Float16 featLDS[TOKTILE * ROWSTR];
    __shared__ float    logitAcc[TOKTILE];

    const int tid = threadIdx.x;
    const int t0  = blockIdx.x * TOKTILE;

    // ---- build 16 x 1120 f16 feature tile in LDS (16 threads per token) ----
    {
        const int tm = tid >> 4;                 // token in tile, 0..15
        const int e  = tid & 15;                 // element lane, 0..15
        const float* xr = x + (size_t)(t0 + tm) * 35;
        _Float16* row = featLDS + tm * ROWSTR;
        float x0 = xr[0], x1 = xr[1];
        row[e]      = (_Float16)x0;  row[16 + e] = (_Float16)x0;   // raw0 broadcast
        row[32 + e] = (_Float16)x1;  row[48 + e] = (_Float16)x1;   // raw1 broadcast
#pragma unroll 1
        for (int f = 0; f < NFEAT; ++f) {
            int idx = (int)xr[2 + f];            // broadcast load within half-wave
            const float* te = tables + ((size_t)f * VDIM + idx) * EDIM;
            row[64 + f * 32 + e]      = (_Float16)te[e];
            row[64 + f * 32 + 16 + e] = (_Float16)te[16 + e];
        }
    }
    if (tid < TOKTILE) logitAcc[tid] = 0.0f;
    __syncthreads();

    // ---- WMMA main loop: 2 N-tiles per wave, double-buffered over K ----
    const int wave = tid >> 5;                   // 0..7
    const int lane = tid & 31;
    const int half = lane >> 4;                  // 0: lanes 0-15, 1: lanes 16-31
    const int l16  = lane & 15;
    const int kA0  = half ? 16 : 0;              // A layout: K {0..15} / {16..31}
    const int kA1  = half ? 24 : 8;

    const _Float16* arow0 = featLDS + l16 * ROWSTR;       // this lane's A row
    const v16h* bbase = ((const v16h*)bpack) + lane;
    const size_t tOffA   = (size_t)(2 * wave) * 32;       // N-tile 2w
    const size_t tOffB   = (size_t)(2 * wave + 1) * 32;   // N-tile 2w+1
    const size_t sStride = (size_t)NTILES * 32;           // per-K-slice stride

    v8f acc0 = {}, acc1 = {};

    // prologue: buffer 0 <- slice 0, buffer 1 <- slice 1
    v8h  alo0 = *(const v8h*)(arow0 + kA0);
    v8h  ahi0 = *(const v8h*)(arow0 + kA1);
    v16h bA0  = bbase[tOffA];
    v16h bB0  = bbase[tOffB];
    v8h  alo1 = *(const v8h*)(arow0 + 32 + kA0);
    v8h  ahi1 = *(const v8h*)(arow0 + 32 + kA1);
    v16h bA1  = bbase[sStride + tOffA];
    v16h bB1  = bbase[sStride + tOffB];

#pragma unroll 1
    for (int s = 0; s < KSLICE - 1; s += 2) {     // slices 0..33, 17 iterations
        // consume buffer 0 (slice s), refill with slice s+2
        {
            v16h a = comb(alo0, ahi0);
            acc0 = __builtin_amdgcn_wmma_f32_16x16x32_f16(
                       false, a, false, bA0, (short)0, acc0, false, false);
            acc1 = __builtin_amdgcn_wmma_f32_16x16x32_f16(
                       false, a, false, bB0, (short)0, acc1, false, false);
        }
        const int s2 = (s + 2 < KSLICE) ? s + 2 : KSLICE - 1;
        alo0 = *(const v8h*)(arow0 + s2 * 32 + kA0);       // ds_load_b128
        ahi0 = *(const v8h*)(arow0 + s2 * 32 + kA1);
        bA0  = bbase[(size_t)s2 * sStride + tOffA];        // global, L2-resident
        bB0  = bbase[(size_t)s2 * sStride + tOffB];

        // consume buffer 1 (slice s+1), refill with slice s+3
        {
            v16h c = comb(alo1, ahi1);
            acc0 = __builtin_amdgcn_wmma_f32_16x16x32_f16(
                       false, c, false, bA1, (short)0, acc0, false, false);
            acc1 = __builtin_amdgcn_wmma_f32_16x16x32_f16(
                       false, c, false, bB1, (short)0, acc1, false, false);
        }
        const int s3 = (s + 3 < KSLICE) ? s + 3 : KSLICE - 1;
        alo1 = *(const v8h*)(arow0 + s3 * 32 + kA0);
        ahi1 = *(const v8h*)(arow0 + s3 * 32 + kA1);
        bA1  = bbase[(size_t)s3 * sStride + tOffA];
        bB1  = bbase[(size_t)s3 * sStride + tOffB];
    }
    // tail: buffer 0 holds slice 34 (via clamped refill)
    {
        v16h a = comb(alo0, ahi0);
        acc0 = __builtin_amdgcn_wmma_f32_16x16x32_f16(
                   false, a, false, bA0, (short)0, acc0, false, false);
        acc1 = __builtin_amdgcn_wmma_f32_16x16x32_f16(
                   false, a, false, bB0, (short)0, acc1, false, false);
    }

    // ---- fused epilogue: +b1, ReLU, *W2[n], reduce both tiles over n ----
    const int n0 = (2 * wave) * 16 + l16;
    const int n1 = n0 + 16;
    const float bias0 = b1[n0], w20 = W2[n0];
    const float bias1 = b1[n1], w21 = W2[n1];
    const int mrow = half ? 8 : 0;
#pragma unroll
    for (int r = 0; r < 8; ++r) {                // D layout: row = r + 8*half, col = l16
        float v0 = acc0[r] + bias0; v0 = v0 > 0.0f ? v0 : 0.0f;
        float v1 = acc1[r] + bias1; v1 = v1 > 0.0f ? v1 : 0.0f;
        atomicAdd(&logitAcc[r + mrow], v0 * w20 + v1 * w21);   // ds_add_f32
    }
    __syncthreads();
    if (tid < TOKTILE) logits[t0 + tid] = logitAcc[tid] + b2[0];
}

// ---------------------------------------------------------------------------
// Kernel 3: softmax over P=100 per batch row; one wave32 per row.
// ---------------------------------------------------------------------------
__global__ __launch_bounds__(32)
void softmax_kernel(const float* __restrict__ logits, float* __restrict__ out)
{
    const int b    = blockIdx.x;
    const int lane = threadIdx.x;
    const float* L = logits + (size_t)b * PDIM;

    float v[4];
    float mx = -INFINITY;
#pragma unroll
    for (int i = 0; i < 4; ++i) {
        int p = lane + i * 32;
        v[i] = (p < PDIM) ? L[p] : -INFINITY;
        mx = fmaxf(mx, v[i]);
    }
#pragma unroll
    for (int d = 16; d >= 1; d >>= 1) mx = fmaxf(mx, __shfl_xor(mx, d, 32));

    float e[4];
    float sum = 0.0f;
#pragma unroll
    for (int i = 0; i < 4; ++i) {
        int p = lane + i * 32;
        e[i] = (p < PDIM) ? __expf(v[i] - mx) : 0.0f;
        sum += e[i];
    }
#pragma unroll
    for (int d = 16; d >= 1; d >>= 1) sum += __shfl_xor(sum, d, 32);

    const float inv = 1.0f / sum;
#pragma unroll
    for (int i = 0; i < 4; ++i) {
        int p = lane + i * 32;
        if (p < PDIM) out[(size_t)b * PDIM + p] = e[i] * inv;
    }
}

// ---------------------------------------------------------------------------
extern "C" void kernel_launch(void* const* d_in, const int* in_sizes, int n_in,
                              void* d_out, int out_size, void* d_ws, size_t ws_size,
                              hipStream_t stream)
{
    const float* x      = (const float*)d_in[0];   // (B,P,35)
    const float* tables = (const float*)d_in[1];   // (NF,V,E)
    const float* W1     = (const float*)d_in[2];   // (H, 1120)
    const float* b1     = (const float*)d_in[3];   // (H)
    const float* W2     = (const float*)d_in[4];   // (1, H)
    const float* b2     = (const float*)d_in[5];   // (1)
    float* out          = (float*)d_out;           // (B,P,1) softmax

    // workspace layout: packed f16 B-fragments, then per-token logits
    const size_t bpackBytes = (size_t)KSLICE * NTILES * 32 * 16 * sizeof(_Float16); // 573,440 B
    _Float16* bpack = (_Float16*)d_ws;
    float*    logit = (float*)((char*)d_ws + bpackBytes);                           // 102,400 B

    // 1) pack W1 into WMMA B-fragment layout (f16)
    {
        const int total = KSLICE * NTILES * 32;    // 17,920
        pack_w1_kernel<<<total / 256, 256, 0, stream>>>(W1, bpack);
    }
    // 2) fused gather + WMMA GEMM + epilogue -> logits
    {
        const int nTok = BDIM * PDIM;              // 25,600
        embed_ffnn_kernel<<<nTok / TOKTILE, 256, 0, stream>>>(
            x, tables, bpack, b1, W2, b2, logit);
    }
    // 3) softmax over positions
    softmax_kernel<<<BDIM, 32, 0, stream>>>(logit, out);
}
